// NAIS_region_distance_Embedding_65240553226373
// MI455X (gfx1250) — matrix operand order repaired and checked by reading, not compiled, not run on hardware
//
#include <hip/hip_runtime.h>
#include <hip/hip_bf16.h>

typedef __attribute__((ext_vector_type(16))) _Float16 v16h;
typedef __attribute__((ext_vector_type(8)))  _Float16 v8h;
typedef __attribute__((ext_vector_type(8)))  float    v8f;

#define B_SZ   2048
#define H_SZ   100
#define D_SZ   128
#define TWO_D  256
#define HID    256
#define MROWS  112      // 100 rows padded to 7 x 16
#define XSTRIDE 264     // 256 + 8 halves pad -> 4-bank row shift, conflict-free

// ---------------------------------------------------------------------------
// Butterfly add across each 16-lane group using DPP16 (pure VALU, no LDS):
//   xor1 -> quad_perm(1,0,3,2)=0xB1, xor2 -> quad_perm(2,3,0,1)=0x4E,
//   xor4 -> row_half_mirror=0x141,   xor8 -> row_mirror=0x140
// ---------------------------------------------------------------------------
__device__ __forceinline__ float row16_reduce_add(float v) {
  v += __builtin_bit_cast(float, __builtin_amdgcn_update_dpp(
           0, __builtin_bit_cast(int, v), 0xB1, 0xF, 0xF, true));
  v += __builtin_bit_cast(float, __builtin_amdgcn_update_dpp(
           0, __builtin_bit_cast(int, v), 0x4E, 0xF, 0xF, true));
  v += __builtin_bit_cast(float, __builtin_amdgcn_update_dpp(
           0, __builtin_bit_cast(int, v), 0x141, 0xF, 0xF, true));
  v += __builtin_bit_cast(float, __builtin_amdgcn_update_dpp(
           0, __builtin_bit_cast(int, v), 0x140, 0xF, 0xF, true));
  return v;
}

// ---------------------------------------------------------------------------
// Kernel 0: pack W1 [256 x 256] f32 into the per-lane WMMA B-fragment layout
// for v_wmma_f32_16x16x32_f16 (lanes 0-15: K 0-7 / 16-23 of column N=lane; lanes
// 16-31: K 8-15 / 24-31). Each lane's 16 halves contiguous -> one 32B load in GEMM.
// ---------------------------------------------------------------------------
__global__ __launch_bounds__(256) void pack_w1_kernel(const float* __restrict__ W1,
                                                      _Float16* __restrict__ W1p) {
  int t    = blockIdx.x * 256 + threadIdx.x;   // 0 .. 65535
  int j    = t & 15;
  int lane = (t >> 4) & 31;
  int kt   = (t >> 9) & 7;
  int nt   = t >> 12;
  int n = nt * 16 + (lane & 15);
  int k = kt * 32 + ((j >> 3) * 16) + ((lane >> 4) * 8) + (j & 7);
  W1p[t] = (_Float16)W1[n * TWO_D + k];
}

// ---------------------------------------------------------------------------
// Main kernel: one workgroup per batch element. 256 threads = 8 wave32.
// ---------------------------------------------------------------------------
__global__ __launch_bounds__(256) void nais_kernel(
    const int*   __restrict__ history,   const int*   __restrict__ target,
    const int*   __restrict__ hreg,      const int*   __restrict__ treg,
    const float* __restrict__ tdist,
    const float* __restrict__ E_hist,    const float* __restrict__ E_tgt,
    const float* __restrict__ E_reg,     const float* __restrict__ E_dist,
    const _Float16* __restrict__ W1p,
    const float* __restrict__ b1,        const float* __restrict__ w2,
    float* __restrict__ out)
{
  __shared__ _Float16 Xs[MROWS * XSTRIDE];   // 59136 B
  __shared__ float tgt_s[TWO_D];             //  1024 B
  __shared__ float scorep[8 * MROWS];        //  3584 B (per-wave score partials)

  const int tid  = threadIdx.x;
  const int b    = blockIdx.x;
  const int lane = tid & 31;
  const int wave = tid >> 5;

  // ---- target embedding concat into LDS ----
  const int titem = target[b];
  const int tr    = treg[b];
  tgt_s[tid] = (tid < D_SZ) ? E_tgt[titem * D_SZ + tid]
                            : E_reg[tr * D_SZ + (tid - D_SZ)];

  // ---- sumE = sum(E_dist[0,:]) via tree reduce (scorep as scratch) ----
  scorep[tid] = (tid < D_SZ) ? E_dist[tid] : 0.f;
  __syncthreads();
  for (int s = 128; s > 0; s >>= 1) {
    if (tid < s) scorep[tid] += scorep[tid + s];
    __syncthreads();
  }
  const float sumE = scorep[0];
  __syncthreads();

  // ---- zero score partials + X pad rows (cols 0..255 of rows 100..111) ----
  for (int i = tid; i < 8 * MROWS; i += 256) scorep[i] = 0.f;
  for (int i = tid; i < (MROWS - H_SZ) * TWO_D; i += 256) {
    int r = H_SZ + (i >> 8);
    int c = i & 255;
    Xs[r * XSTRIDE + c] = (_Float16)0.f;
  }
  __syncthreads();

  // ---- phase 1: gather hist rows, x = hist*tgt -> f16 LDS ----
  // 2 rows x (2 cols/thread) per iteration: b64 global loads, b32 LDS stores.
  {
    const int sub = tid >> 7;            // which row of the pair
    const int c2  = (tid & 127) * 2;     // column base (even)
    const float t0 = tgt_s[c2];
    const float t1 = tgt_s[c2 + 1];
    for (int hh = 0; hh < H_SZ; hh += 2) {
      int h    = hh + sub;
      int item = history[b * H_SZ + h];
      int rg   = hreg[b * H_SZ + h];
      const float* src = (c2 < D_SZ) ? &E_hist[item * D_SZ + c2]
                                     : &E_reg[rg * D_SZ + (c2 - D_SZ)];
      // prefetch next pair's row (global_prefetch_b8) to hide gather latency
      if (hh + 2 < H_SZ) {
        int h2   = hh + 2 + sub;
        int it2  = history[b * H_SZ + h2];
        int rg2  = hreg[b * H_SZ + h2];
        const float* p2 = (c2 < D_SZ) ? &E_hist[it2 * D_SZ + c2]
                                      : &E_reg[rg2 * D_SZ + (c2 - D_SZ)];
        __builtin_prefetch(p2, 0, 0);
      }
      float2 v = *(const float2*)src;
      union { _Float16 h[2]; unsigned u; } p;
      p.h[0] = (_Float16)(v.x * t0);
      p.h[1] = (_Float16)(v.y * t1);
      *(unsigned*)&Xs[h * XSTRIDE + c2] = p.u;
    }
  }
  __syncthreads();

  // ---- phase 2: WMMA GEMM [112x256]x[256x256] + fused relu/b1/w2 epilogue ----
  {
    const int halfsel = lane >> 4;   // 0 or 1
    const int lan15   = lane & 15;
    for (int ntl = 0; ntl < 2; ++ntl) {
      const int nt  = wave * 2 + ntl;          // this wave's N-tile
      const int n   = nt * 16 + lan15;
      const float b1v = b1[n];
      const float w2v = w2[n];
      for (int mt = 0; mt < 7; ++mt) {
        v8f c = {};
        const int m = mt * 16 + lan15;         // A-fragment row for this lane
        #pragma unroll
        for (int kt = 0; kt < 8; ++kt) {
          // A (16x32 f16): lanes 0-15 hold K 0-7 / 16-23, lanes 16-31 K 8-15 / 24-31
          union { v16h v; v8h h[2]; } au;
          const _Float16* ap = &Xs[m * XSTRIDE + kt * 32 + halfsel * 8];
          au.h[0] = *(const v8h*)(ap);
          au.h[1] = *(const v8h*)(ap + 16);
          // B: contiguous packed fragment, 32B per lane (L2-resident)
          v16h bv = *((const v16h*)W1p + ((nt * 8 + kt) * 32 + lane));
          c = __builtin_amdgcn_wmma_f32_16x16x32_f16(
                  false, au.v, false, bv, (short)0, c, false, false);
        }
        // epilogue: relu(c + b1) * w2, DPP butterfly over the 16 N-lanes,
        // unconditional same-value LDS accumulate (wave-private region).
        #pragma unroll
        for (int j = 0; j < 8; ++j) {
          float v = fmaxf(c[j] + b1v, 0.f) * w2v;
          v = row16_reduce_add(v);               // all 16 lanes now hold sum
          int mm = mt * 16 + halfsel * 8 + j;    // C layout: VGPR j -> row
          float* sp = &scorep[wave * MROWS + mm];
          *sp += v;  // 16 lanes write identical value; 0-15 vs 16-31 disjoint
        }
      }
    }
  }
  __syncthreads();

  // ---- phase 3: dot from Xs, mask/exp, beta=0.5 normalize, pred, sigmoid ----
  float e = 0.f, dv = 0.f;
  if (tid < H_SZ) {
    // dot[h] = sum_d x[h,d] straight from the f16 tile in LDS
    const v8h* rp = (const v8h*)&Xs[tid * XSTRIDE];
    #pragma unroll
    for (int i = 0; i < 32; ++i) {
      v8h q = rp[i];
      #pragma unroll
      for (int j = 0; j < 8; ++j) dv += (float)q[j];
    }
    float s = 0.f;
    #pragma unroll
    for (int w = 0; w < 8; ++w) s += scorep[w * MROWS + tid];
    float td = tdist[b];
    e = expf(s + td * sumE);
    if (history[b * H_SZ + tid] == target[b]) e = 0.f;
  }
  __syncthreads();
  tgt_s[tid]  = e;          // reuse as reduction buffers
  scorep[tid] = e * dv;
  __syncthreads();
  for (int s = 128; s > 0; s >>= 1) {
    if (tid < s) {
      tgt_s[tid]  += tgt_s[tid + s];
      scorep[tid] += scorep[tid + s];
    }
    __syncthreads();
  }
  if (tid == 0) {
    float pred = scorep[0] / sqrtf(tgt_s[0]);   // beta = 0.5
    out[b] = 1.f / (1.f + expf(-pred));
  }
}

// ---------------------------------------------------------------------------
extern "C" void kernel_launch(void* const* d_in, const int* in_sizes, int n_in,
                              void* d_out, int out_size, void* d_ws, size_t ws_size,
                              hipStream_t stream) {
  const int*   history = (const int*)  d_in[0];
  const int*   target  = (const int*)  d_in[1];
  const int*   hregion = (const int*)  d_in[2];
  const int*   tregion = (const int*)  d_in[3];
  const float* tdist   = (const float*)d_in[4];
  const float* E_hist  = (const float*)d_in[5];
  const float* E_tgt   = (const float*)d_in[6];
  const float* E_reg   = (const float*)d_in[7];
  const float* E_dist  = (const float*)d_in[8];
  const float* W1      = (const float*)d_in[9];
  const float* b1      = (const float*)d_in[10];
  const float* w2      = (const float*)d_in[11];
  float*       outp    = (float*)d_out;

  _Float16* W1p = (_Float16*)d_ws;   // 65536 halves = 128 KB packed W1

  pack_w1_kernel<<<HID * TWO_D / 256, 256, 0, stream>>>(W1, W1p);
  nais_kernel<<<B_SZ, 256, 0, stream>>>(history, target, hregion, tregion, tdist,
                                        E_hist, E_tgt, E_reg, E_dist,
                                        W1p, b1, w2, outp);
}
